// NormPreserveDeformConvPack_56504589746699
// MI455X (gfx1250) — matrix operand order repaired and checked by reading, not compiled, not run on hardware
//
#include <hip/hip_runtime.h>
#include <hip/hip_bf16.h>

#define BSZ   8
#define CIN   128
#define COUT  128
#define HH    64
#define WW    64
#define KTAPS 9
#define HO    64
#define WO    64
#define EPSF  1e-4f
#define RTOT  1152          // CIN * KTAPS
#define NKSTEP 36           // RTOT / 32
#define NMTILE 8            // COUT / 16

typedef __bf16 v16bf __attribute__((ext_vector_type(16)));
typedef float  v8f   __attribute__((ext_vector_type(8)));

__device__ __forceinline__ unsigned short f32_to_bf16_rne(float f) {
  unsigned int u = __builtin_bit_cast(unsigned int, f);
  unsigned int r = u + 0x7FFFu + ((u >> 16) & 1u);   // round-to-nearest-even
  return (unsigned short)(r >> 16);
}

// ---------------------------------------------------------------------------
// Kernel 1: norm-preserving weight normalization.
// wn[o,c,k] = w[o,c,k] / (sqrt(sum_c w^2 + CIN*EPS) * sqrt(9))
// Output layout wn2[o][r], r = k*CIN + c (k-major so a 32-wide K-step shares k).
// ---------------------------------------------------------------------------
__global__ void wnorm_kernel(const float* __restrict__ weight,
                             float* __restrict__ wn2) {
  __shared__ float ssum[KTAPS];
  const int o = blockIdx.x;
  const int c = threadIdx.x;
  if (c < KTAPS) ssum[c] = 0.0f;
  __syncthreads();
  float w[KTAPS];
#pragma unroll
  for (int k = 0; k < KTAPS; ++k) {
    w[k] = weight[(o * CIN + c) * KTAPS + k];
    atomicAdd(&ssum[k], w[k] * w[k]);            // ds_add_f32
  }
  __syncthreads();
#pragma unroll
  for (int k = 0; k < KTAPS; ++k) {
    float len = sqrtf(ssum[k] + (float)CIN * EPSF);
    wn2[(size_t)o * RTOT + k * CIN + c] = w[k] / (len * 3.0f);
  }
}

// ---------------------------------------------------------------------------
// Kernel 2: pack normalized weights into the CDNA5 16-bit A-matrix (16x32)
// per-lane fragment layout, so the GEMM loads A with coalesced b128 loads.
// Layout per (mtile, kstep): 32 lanes x 8 dwords (2 bf16 each).
//   lane L: M = mtile*16 + (L&15), Kbase = (L>>4)*8
//   dword j<4 : K = Kbase + 2j (+1) ; j>=4 : K = 16 + Kbase + 2(j-4) (+1)
// ---------------------------------------------------------------------------
__global__ void wpack_kernel(const float* __restrict__ wn2,
                             unsigned int* __restrict__ packedA) {
  const int bk    = blockIdx.x;         // 0..287
  const int mtile = bk / NKSTEP;
  const int kstep = bk % NKSTEP;
  const int lane  = threadIdx.x;        // 0..31
  const int M     = mtile * 16 + (lane & 15);
  const int kb    = (lane >> 4) * 8;
  unsigned int* dst = packedA + ((size_t)(mtile * NKSTEP + kstep) * 32 + lane) * 8;
#pragma unroll
  for (int j = 0; j < 8; ++j) {
    int k0 = (j < 4) ? (kb + 2 * j) : (kb + 16 + 2 * (j - 4));
    int r0 = kstep * 32 + k0;
    unsigned short lo = f32_to_bf16_rne(wn2[(size_t)M * RTOT + r0]);
    unsigned short hi = f32_to_bf16_rne(wn2[(size_t)M * RTOT + r0 + 1]);
    dst[j] = (unsigned int)lo | ((unsigned int)hi << 16);
  }
}

// ---------------------------------------------------------------------------
// Kernel 3: deformable im2col + WMMA GEMM.
// Block = 256 threads (8 wave32) handles full COUT=128 x 32 pixels.
// Wave m: M-tile m (16 couts) x 2 N-tiles (16 px each). 36 K-steps of 32.
// ---------------------------------------------------------------------------
__launch_bounds__(256, 2)
__global__ void dconv_kernel(const float* __restrict__ x,
                             const float* __restrict__ offs,
                             const unsigned int* __restrict__ packedA,
                             float* __restrict__ out) {
  __shared__ float s_tw[KTAPS * 32][4];
  __shared__ int   s_tix[KTAPS * 32][4];
  __shared__ __align__(16) unsigned int s_bfrag[2][32][8];  // B-fragment layout

  const int bid = blockIdx.x;
  const int wt  = bid & 1;           // which 32-pixel half of the row
  const int ho  = (bid >> 1) & 63;
  const int b   = bid >> 7;
  const int wo_base = wt * 32;

  const int tid  = threadIdx.x;
  const int lane = tid & 31;
  const int wv   = tid >> 5;

  // ---- per-(pixel,k) bilinear tap precompute (reused across all 128 c) ----
  for (int t = tid; t < KTAPS * 32; t += 256) {
    int k  = t >> 5;
    int p  = t & 31;
    int wo = wo_base + p;
    int ky = k / 3, kx = k - 3 * ky;
    float offy = offs[(((size_t)b * 18 + 2 * k    ) * HO + ho) * WO + wo];
    float offx = offs[(((size_t)b * 18 + 2 * k + 1) * HO + ho) * WO + wo];
    float py = (float)(ho - 1 + ky) + offy;
    float px = (float)(wo - 1 + kx) + offx;
    float y0 = floorf(py), x0 = floorf(px);
    float ly = py - y0,    lx = px - x0;
#pragma unroll
    for (int cn = 0; cn < 4; ++cn) {
      int dy = cn >> 1, dx = cn & 1;
      float yy = y0 + (float)dy, xx = x0 + (float)dx;
      bool valid = (yy >= 0.0f) && (yy < (float)HH) && (xx >= 0.0f) && (xx < (float)WW);
      float wgt = (dy ? ly : 1.0f - ly) * (dx ? lx : 1.0f - lx);
      int yi = min(max((int)yy, 0), HH - 1);
      int xi = min(max((int)xx, 0), WW - 1);
      s_tw[t][cn]  = valid ? wgt : 0.0f;
      s_tix[t][cn] = yi * WW + xi;
    }
  }
  __syncthreads();

  const float* xb = x + (size_t)b * CIN * HH * WW;
  const uint4* pA = (const uint4*)packedA;

  v8f acc0 = {0.f, 0.f, 0.f, 0.f, 0.f, 0.f, 0.f, 0.f};
  v8f acc1 = {0.f, 0.f, 0.f, 0.f, 0.f, 0.f, 0.f, 0.f};

  for (int kstep = 0; kstep < NKSTEP; ++kstep) {
    const int k     = kstep >> 2;
    const int cbase = (kstep & 3) * 32;

    // A fragment: two coalesced b128 loads (pre-swizzled in kernel 2)
    union { uint4 q[2]; v16bf v; } afrag;
    int aidx  = ((wv * NKSTEP + kstep) * 32 + lane) * 2;
    afrag.q[0] = pA[aidx];
    afrag.q[1] = pA[aidx + 1];

    // Build B tile (32 r x 32 px, bf16) straight into B-fragment layout.
    // lane -> pixel p, wave wv -> rows rr = wv + 8*it (channel c = cbase+rr).
    const int p = lane;
    const int t = k * 32 + p;
    float w0 = s_tw[t][0], w1 = s_tw[t][1], w2 = s_tw[t][2], w3 = s_tw[t][3];
    int   i0 = s_tix[t][0], i1 = s_tix[t][1], i2 = s_tix[t][2], i3 = s_tix[t][3];
#pragma unroll
    for (int it = 0; it < 4; ++it) {
      int rr = wv + it * 8;
      const float* xc = xb + (size_t)(cbase + rr) * (HH * WW);
      float val = w0 * xc[i0] + w1 * xc[i1] + w2 * xc[i2] + w3 * xc[i3];
      unsigned short h = f32_to_bf16_rne(val);
      // B layout (K x N, 16-bit): lane = (N&15) + 16*((K/8)&1); dword = (K%8)/2 + 4*(K/16)
      int nt   = p >> 4;
      int lb   = (p & 15) + (((rr >> 3) & 1) << 4);
      int js   = ((rr & 7) >> 1) + ((rr >> 4) << 2);
      int half = rr & 1;
      ((unsigned short*)&s_bfrag[nt][lb][js])[half] = h;    // ds_store_b16
    }
    __syncthreads();

    union { uint4 q[2]; v16bf v; } bf0, bf1;
    const uint4* b0p = (const uint4*)&s_bfrag[0][lane][0];
    const uint4* b1p = (const uint4*)&s_bfrag[1][lane][0];
    bf0.q[0] = b0p[0]; bf0.q[1] = b0p[1];                   // ds_load_b128 x2
    bf1.q[0] = b1p[0]; bf1.q[1] = b1p[1];

    acc0 = __builtin_amdgcn_wmma_f32_16x16x32_bf16(false, afrag.v, false, bf0.v,
                                                   (short)0, acc0, false, false);
    acc1 = __builtin_amdgcn_wmma_f32_16x16x32_bf16(false, afrag.v, false, bf1.v,
                                                   (short)0, acc1, false, false);
    __syncthreads();
  }

  // ---- epilogue: C layout -> (b, cout, ho, wo), 64B-coalesced row stores ----
  const int nrow = lane & 15;
  const int moff = (lane >> 4) * 8;
#pragma unroll
  for (int j = 0; j < 8; ++j) {
    int cout = wv * 16 + j + moff;
    size_t base = (((size_t)b * COUT + cout) * HO + ho) * WO + wo_base;
    out[base + nrow]      = acc0[j];
    out[base + 16 + nrow] = acc1[j];
  }
}

// ---------------------------------------------------------------------------
extern "C" void kernel_launch(void* const* d_in, const int* in_sizes, int n_in,
                              void* d_out, int out_size, void* d_ws, size_t ws_size,
                              hipStream_t stream) {
  (void)in_sizes; (void)n_in; (void)out_size; (void)ws_size;
  const float* x      = (const float*)d_in[0];
  const float* offset = (const float*)d_in[1];
  const float* weight = (const float*)d_in[2];
  float* out = (float*)d_out;

  unsigned int* packedA = (unsigned int*)d_ws;                  // 294,912 B
  float*        wn2     = (float*)((char*)d_ws + 294912);       // 589,824 B

  wnorm_kernel<<<COUT, CIN, 0, stream>>>(weight, wn2);
  wpack_kernel<<<NMTILE * NKSTEP, 32, 0, stream>>>(wn2, packedA);
  dconv_kernel<<<BSZ * HO * (WO / 32), 256, 0, stream>>>(x, offset, packedA, out);
}
